// DeepBPNetModule_33809982554515
// MI455X (gfx1250) — matrix operand order repaired and checked by reading, not compiled, not run on hardware
//
#include <hip/hip_runtime.h>
#include <math.h>
#include <stdint.h>

typedef __attribute__((ext_vector_type(16))) _Float16 v16h;
typedef __attribute__((ext_vector_type(8)))  float    v8f;

static constexpr int H  = 128;
static constexpr int W  = 128;
static constexpr int HW = H * W;
static constexpr int ND = 16;   // depth hypotheses
static constexpr int DHW = ND * HW;

// ---------------------------------------------------------------------------
// WMMA fragment coordinate maps (CDNA5 wave32, 16x16x32 f16 -> f32)
// A (MxK = 16x32): lanes 0-15 hold M=lane; VGPR0-3 = K 0..7, VGPR4-7 = K 16..23.
//                  lanes 16-31 hold M=lane-16; VGPR0-3 = K 8..15, VGPR4-7 = K 24..31.
// B (KxN = 32x16): same K order as A; lanes 0-15 -> N=lane, lanes 16-31 -> N=lane-16.
// C/D (16x16 f32): lanes 0-15 -> N=lane, M=r;  lanes 16-31 -> N=lane-16, M=r+8.
// ---------------------------------------------------------------------------
__device__ __forceinline__ void a_coord(int lane, int e, int& m, int& k) {
    int half = lane >> 4;
    m = lane & 15;
    int v = e >> 1, hh = e & 1;
    int kk = ((v < 4) ? 0 : 16) + ((v & 3) << 1) + hh;
    k = kk + (half ? 8 : 0);
}
__device__ __forceinline__ void b_coord(int lane, int e, int& k, int& n) {
    n = lane & 15;
    k = ((lane >> 4) ? 8 : 0) + ((e & 8) << 1) + (e & 7);
}
__device__ __forceinline__ v8f wmma_f16(v16h a, v16h b, v8f c) {
    return __builtin_amdgcn_wmma_f32_16x16x32_f16(false, a, false, b, (short)0, c,
                                                  false, false);
}
__device__ __forceinline__ float sigmoid_fast(float x) {
    return __fdividef(1.f, 1.f + __expf(-x));
}

// Async copy one b32 per lane from global to LDS (CDNA5 async path, ASYNCcnt).
// lds_off is the byte offset inside the workgroup's LDS allocation.
__device__ __forceinline__ void async_ld_b32(unsigned lds_off, unsigned long long gaddr) {
    asm volatile("global_load_async_to_lds_b32 %0, %1, off"
                 :: "v"(lds_off), "v"(gaddr) : "memory");
}
__device__ __forceinline__ void wait_async0() {
    asm volatile("s_wait_asynccnt 0x0" ::: "memory");
}

// ---------------------------------------------------------------------------
// Generic IC=32 conv as implicit GEMM. One wave = one (octile, 16-pixel strip).
// Weight index = oc*wOcS + ic*wIcS + wBase + ky*wKyS + kx*wKxS  (handles the
// mw1_w[:, :32, 1, :] row/col slices without repacking).
// ---------------------------------------------------------------------------
template <int KH, int KW, int ACT>  // ACT: 0 = none, 1 = relu
__global__ void conv32_wmma(const float* __restrict__ in, const float* __restrict__ w,
                            const float* __restrict__ bias, float* __restrict__ out,
                            int OT, int wOcS, int wIcS, int wBase, int wKyS, int wKxS) {
    int lane = threadIdx.x & 31, wave = threadIdx.x >> 5;
    int tile = blockIdx.x * 4 + wave;
    int octile = tile % OT;
    int strip  = tile / OT;
    int p0 = strip * 16, y = p0 >> 7, x0 = p0 & 127;
    v8f acc = {};
    for (int t = 0; t < KH * KW; ++t) {
        int ky = t / KW, kx = t % KW;
        int dy = ky - KH / 2, dx = kx - KW / 2;
        v16h a, b;
#pragma unroll
        for (int e = 0; e < 16; ++e) {
            int m, k;
            a_coord(lane, e, m, k);
            a[e] = (_Float16)w[(octile * 16 + m) * wOcS + k * wIcS + wBase +
                               ky * wKyS + kx * wKxS];
            int kb, n;
            b_coord(lane, e, kb, n);
            int iy = y + dy, ix = x0 + n + dx;
            float bv = (iy >= 0 && iy < H && ix >= 0 && ix < W)
                           ? in[kb * HW + iy * W + ix] : 0.f;
            b[e] = (_Float16)bv;
        }
        acc = wmma_f16(a, b, acc);
    }
#pragma unroll
    for (int r = 0; r < 8; ++r) {
        int m = r + ((lane >= 16) ? 8 : 0), n = lane & 15;
        int oc = octile * 16 + m;
        float v = acc[r] + bias[oc];
        if (ACT == 1) v = fmaxf(v, 0.f);
        out[oc * HW + p0 + n] = v;
    }
}

// ---------------------------------------------------------------------------
// dw2 (32->1 1x1) + sigmoid, fused with cost = cost_volume * dwt.
// ---------------------------------------------------------------------------
__global__ void dw2_cost_kernel(const float* __restrict__ hid32,
                                const float* __restrict__ dw2w,
                                const float* __restrict__ dw2b,
                                const float* __restrict__ cv,
                                float* __restrict__ cost) {
    int lane = threadIdx.x & 31, wave = threadIdx.x >> 5;
    int strip = blockIdx.x * 4 + wave;
    int p0 = strip * 16;
    v16h a, b;
    v8f acc = {};
#pragma unroll
    for (int e = 0; e < 16; ++e) {
        int m, k;
        a_coord(lane, e, m, k);
        a[e] = (m == 0) ? (_Float16)dw2w[k] : (_Float16)0.f;
        int kb, n;
        b_coord(lane, e, kb, n);
        b[e] = (_Float16)hid32[kb * HW + p0 + n];
    }
    acc = wmma_f16(a, b, acc);
    if (lane < 16) {
        float dwt = sigmoid_fast(acc[0] + dw2b[0]);
        int p = p0 + lane;
        for (int d = 0; d < ND; ++d) cost[d * HW + p] = cv[d * HW + p] * dwt;
    }
}

// ---------------------------------------------------------------------------
// Smoothness weight map: sigmoid(conv1x1(relu(conv(concat(feat, shift(feat))))))
// conv1: IC=64 (ic>=32 reads clamped-shifted feature), OC=32, 1x3 or 3x1 slice
// of sw1_w. Fused with conv2 (32->1) through LDS. 2 waves per 16-pixel strip.
// ---------------------------------------------------------------------------
__global__ void smooth_kernel(const float* __restrict__ feat,
                              const float* __restrict__ sw1w,
                              const float* __restrict__ sw1b,
                              const float* __restrict__ sw2w,
                              const float* __restrict__ sw2b,
                              float* __restrict__ out, int rowflag, int dy, int dx) {
    __shared__ float hid[32 * 16];
    int lane = threadIdx.x & 31, octile = threadIdx.x >> 5;
    int p0 = blockIdx.x * 16, y = p0 >> 7, x0 = p0 & 127;
    v8f acc = {};
    for (int c6 = 0; c6 < 6; ++c6) {  // K = 64 ic * 3 taps = 192 = 6 chunks
        v16h a, b;
#pragma unroll
        for (int e = 0; e < 16; ++e) {
            int m, k;
            a_coord(lane, e, m, k);
            k += c6 * 32;
            int ic = k / 3, t = k % 3;
            int ky = rowflag ? 1 : t, kx = rowflag ? t : 1;
            a[e] = (_Float16)sw1w[((octile * 16 + m) * 64 + ic) * 9 + ky * 3 + kx];
            int kb, n;
            b_coord(lane, e, kb, n);
            kb += c6 * 32;
            int icb = kb / 3, tb = kb % 3;
            int ty = rowflag ? y : (y + tb - 1);
            int tx = rowflag ? (x0 + n + tb - 1) : (x0 + n);
            float v = 0.f;
            if (ty >= 0 && ty < H && tx >= 0 && tx < W) {
                if (icb < 32) {
                    v = feat[icb * HW + ty * W + tx];
                } else {
                    int cy = min(max(ty + dy, 0), H - 1);
                    int cx = min(max(tx + dx, 0), W - 1);
                    v = feat[(icb - 32) * HW + cy * W + cx];
                }
            }
            b[e] = (_Float16)v;
        }
        acc = wmma_f16(a, b, acc);
    }
#pragma unroll
    for (int r = 0; r < 8; ++r) {
        int m = r + ((lane >= 16) ? 8 : 0), n = lane & 15;
        int oc = octile * 16 + m;
        hid[oc * 16 + n] = fmaxf(acc[r] + sw1b[oc], 0.f);
    }
    __syncthreads();
    if (octile == 0) {
        v16h a, b;
        v8f acc2 = {};
#pragma unroll
        for (int e = 0; e < 16; ++e) {
            int m, k;
            a_coord(lane, e, m, k);
            a[e] = (m == 0) ? (_Float16)sw2w[k] : (_Float16)0.f;
            int kb, n;
            b_coord(lane, e, kb, n);
            b[e] = (_Float16)hid[kb * 16 + n];
        }
        acc2 = wmma_f16(a, b, acc2);
        if (lane < 16) out[p0 + lane] = sigmoid_fast(acc2[0] + sw2b[0]);
    }
}

// ---------------------------------------------------------------------------
// Fused message pass for one (d, line). fpre[:, line] (64x128 f32, strided in
// the column case) is staged into LDS with CDNA5 async global->LDS loads,
// overlapped with the VALU staging of s = cost + pa + pb and the weights.
// Then h = relu(fpre + conv1d(s, w1)) in LDS, m = conv1d(h, w2) + b, and the
// shifted, smoothness-gated next messages are written directly.
// rowflag=1: horizontal lines (m_v path), rowflag=0: vertical lines (m_h).
// ---------------------------------------------------------------------------
__global__ void msg_kernel(const float* __restrict__ cost,
                           const float* __restrict__ pa, const float* __restrict__ pb,
                           const float* __restrict__ fpre,
                           const float* __restrict__ mw1, const float* __restrict__ mw2,
                           const float* __restrict__ mw2b,
                           const float* __restrict__ swa, const float* __restrict__ swb,
                           float* __restrict__ outa, float* __restrict__ outb,
                           int rowflag) {
    // Single LDS object => s_fpre sits at LDS byte offset 0 (async VDST offsets).
    __shared__ float smem[64 * 128 + 64 * (W + 2) + (W + 2) + 192 + 192];
    float* s_fpre = smem;                       // 64*128, async-filled
    float* s_h    = smem + 64 * 128;            // 64*(W+2)
    float* s_s    = s_h + 64 * (W + 2);         // W+2
    float* s_w1   = s_s + (W + 2);              // 64*3
    float* s_w2   = s_w1 + 192;                 // 64*3

    int t = threadIdx.x;              // position along line (0..127)
    int d = blockIdx.x >> 7;
    int line = blockIdx.x & 127;
    int pix = rowflag ? (line * W + t) : (t * W + line);
    int base = d * HW;

    // --- kick off async global->LDS staging of fpre[:, line] ---------------
    {
        unsigned long long gbase =
            (unsigned long long)(uintptr_t)fpre + (unsigned long long)pix * 4ull;
        for (int c = 0; c < 64; ++c) {
            unsigned lds_off = (unsigned)((c * 128 + t) * 4);
            async_ld_b32(lds_off, gbase + (unsigned long long)c * (HW * 4ull));
        }
    }
    __builtin_prefetch(&swa[pix], 0, 1);   // global_prefetch_b8
    __builtin_prefetch(&swb[pix], 0, 1);

    // --- overlap: weights + s line via the normal load path ----------------
    if (t < 64) {
        for (int j = 0; j < 3; ++j) {
            int ky = rowflag ? 1 : j, kx = rowflag ? j : 1;
            s_w1[t * 3 + j] = mw1[(t * 33 + 32) * 9 + ky * 3 + kx];  // mw1_s slice
            s_w2[t * 3 + j] = mw2[t * 9 + ky * 3 + kx];              // mw2 slice
        }
    }
    s_s[t + 1] = cost[base + pix] + pa[base + pix] + pb[base + pix];
    if (t == 0) { s_s[0] = 0.f; s_s[W + 1] = 0.f; }

    wait_async0();        // this wave's async loads have landed in LDS
    __syncthreads();      // make all waves' staging visible

    for (int c = 0; c < 64; ++c) {
        float h = s_fpre[c * 128 + t] + s_s[t] * s_w1[c * 3] +
                  s_s[t + 1] * s_w1[c * 3 + 1] + s_s[t + 2] * s_w1[c * 3 + 2];
        s_h[c * (W + 2) + t + 1] = fmaxf(h, 0.f);
    }
    if (t < 64) { s_h[t * (W + 2)] = 0.f; s_h[t * (W + 2) + W + 1] = 0.f; }
    __syncthreads();
    float m = mw2b[0];
    for (int c = 0; c < 64; ++c) {
        const float* hr = &s_h[c * (W + 2)];
        m += hr[t] * s_w2[c * 3] + hr[t + 1] * s_w2[c * 3 + 1] +
             hr[t + 2] * s_w2[c * 3 + 2];
    }
    float wa = swa[pix], wb = swb[pix];
    if (rowflag) {
        int y = line;
        if (y > 0)     outa[base + (y - 1) * W + t] = m * wa;
        else           outa[base + (H - 1) * W + t] = 0.f;   // mu_n[H-1] = 0
        if (y < H - 1) outb[base + (y + 1) * W + t] = m * wb;
        else           outb[base + t] = 0.f;                 // md_n[0] = 0
    } else {
        int x = line;
        if (x > 0)     outa[base + t * W + (x - 1)] = m * wa;
        else           outa[base + t * W + (W - 1)] = 0.f;   // ml_n[W-1] = 0
        if (x < W - 1) outb[base + t * W + (x + 1)] = m * wb;
        else           outb[base + t * W] = 0.f;             // mr_n[0] = 0
    }
}

// ---------------------------------------------------------------------------
// Belief head per d-slice: hid = relu(conv3x3(5->64)) on a 3x18 halo tile via
// WMMA (K = 5*9 = 45 padded to 64), staged in LDS, then conv3x3(64->1).
// xin = [mu|md|ml|mr|cost], channel stride DHW. One block = one 16-pixel strip.
// ---------------------------------------------------------------------------
__global__ void belief_kernel(const float* __restrict__ xin,
                              const float* __restrict__ bw1,
                              const float* __restrict__ bw1b,
                              const float* __restrict__ bw2,
                              const float* __restrict__ bw2b,
                              float* __restrict__ out) {
    __shared__ float s_hid[64 * 54];
    int lane = threadIdx.x & 31, octile = threadIdx.x >> 5;
    int d = blockIdx.x >> 10;
    int strip = blockIdx.x & 1023;
    int p0 = strip * 16, y = p0 >> 7, x0 = p0 & 127;
    for (int pt = 0; pt < 4; ++pt) {  // 54 hid pixels (3 rows x 18) in 4 tiles
        v8f acc = {};
        for (int kc = 0; kc < 2; ++kc) {
            v16h a, b;
#pragma unroll
            for (int e = 0; e < 16; ++e) {
                int m, k;
                a_coord(lane, e, m, k);
                k += kc * 32;
                float av = 0.f;
                if (k < 45) {
                    int ic = k / 9, tap = k % 9;
                    av = bw1[(octile * 16 + m) * 45 + ic * 9 + tap];
                }
                a[e] = (_Float16)av;
                int kb, n;
                b_coord(lane, e, kb, n);
                kb += kc * 32;
                float bv = 0.f;
                int q = pt * 16 + n;
                if (kb < 45 && q < 54) {
                    int ic = kb / 9, tap = kb % 9;
                    int r = q / 18, cx = q % 18;
                    int iy = y - 1 + r, ix = x0 - 1 + cx;
                    int yy = iy + tap / 3 - 1, xx = ix + tap % 3 - 1;
                    if (yy >= 0 && yy < H && xx >= 0 && xx < W)
                        bv = xin[ic * DHW + d * HW + yy * W + xx];
                }
                b[e] = (_Float16)bv;
            }
            acc = wmma_f16(a, b, acc);
        }
#pragma unroll
        for (int r = 0; r < 8; ++r) {
            int m = r + ((lane >= 16) ? 8 : 0), n = lane & 15;
            int q = pt * 16 + n;
            if (q < 54) {
                int rr = q / 18, cx = q % 18;
                int iy = y - 1 + rr, ix = x0 - 1 + cx;
                int oc = octile * 16 + m;
                float v = 0.f;  // hid is zero-padded outside the image
                if (iy >= 0 && iy < H && ix >= 0 && ix < W)
                    v = fmaxf(acc[r] + bw1b[oc], 0.f);
                s_hid[oc * 54 + q] = v;
            }
        }
    }
    __syncthreads();
    if (threadIdx.x < 16) {
        int j = threadIdx.x, x = x0 + j;
        float o = bw2b[0];
        for (int c = 0; c < 64; ++c)
#pragma unroll
            for (int ky = 0; ky < 3; ++ky)
#pragma unroll
                for (int kx = 0; kx < 3; ++kx)
                    o += s_hid[c * 54 + ky * 18 + j + kx] * bw2[c * 9 + ky * 3 + kx];
        out[d * HW + y * W + x] = o;
    }
}

__global__ void zero_kernel(float* __restrict__ p, int n) {
    int i = blockIdx.x * blockDim.x + threadIdx.x;
    if (i < n) p[i] = 0.f;
}

// ---------------------------------------------------------------------------
extern "C" void kernel_launch(void* const* d_in, const int* in_sizes, int n_in,
                              void* d_out, int out_size, void* d_ws, size_t ws_size,
                              hipStream_t stream) {
    (void)in_sizes; (void)n_in; (void)out_size; (void)ws_size;
    const float* feature = (const float*)d_in[0];
    const float* cv      = (const float*)d_in[1];
    const float* dw1w = (const float*)d_in[2];
    const float* dw1b = (const float*)d_in[3];
    const float* dw2w = (const float*)d_in[4];
    const float* dw2b = (const float*)d_in[5];
    const float* sw1w = (const float*)d_in[6];
    const float* sw1b = (const float*)d_in[7];
    const float* sw2w = (const float*)d_in[8];
    const float* sw2b = (const float*)d_in[9];
    const float* mw1w = (const float*)d_in[10];
    const float* mw1b = (const float*)d_in[11];
    const float* mw2w = (const float*)d_in[12];
    const float* mw2b = (const float*)d_in[13];
    const float* bw1w = (const float*)d_in[14];
    const float* bw1b = (const float*)d_in[15];
    const float* bw2w = (const float*)d_in[16];
    const float* bw2b = (const float*)d_in[17];

    float* ws = (float*)d_ws;
    float* hid32 = ws;  ws += 32 * HW;
    float* fprer = ws;  ws += 64 * HW;
    float* fprec = ws;  ws += 64 * HW;
    float* sw_u  = ws;  ws += HW;
    float* sw_d  = ws;  ws += HW;
    float* sw_l  = ws;  ws += HW;
    float* sw_r  = ws;  ws += HW;
    float* bufA  = ws;  ws += 4 * DHW;
    float* bufB  = ws;  ws += 4 * DHW;   // final messages land here...
    float* cost  = ws;  ws += DHW;       // ...with cost contiguous => 5-channel

    // dwt path: 32->32 3x3 relu, then 32->1 1x1 sigmoid * cost_volume
    conv32_wmma<3, 3, 1><<<512, 128, 0, stream>>>(feature, dw1w, dw1b, hid32,
                                                  2, 288, 9, 0, 3, 1);
    dw2_cost_kernel<<<256, 128, 0, stream>>>(hid32, dw2w, dw2b, cv, cost);

    // smoothness maps (virtual concat + clamped shift)
    smooth_kernel<<<1024, 64, 0, stream>>>(feature, sw1w, sw1b, sw2w, sw2b, sw_u, 1, -1, 0);
    smooth_kernel<<<1024, 64, 0, stream>>>(feature, sw1w, sw1b, sw2w, sw2b, sw_d, 1,  1, 0);
    smooth_kernel<<<1024, 64, 0, stream>>>(feature, sw1w, sw1b, sw2w, sw2b, sw_l, 0, 0, -1);
    smooth_kernel<<<1024, 64, 0, stream>>>(feature, sw1w, sw1b, sw2w, sw2b, sw_r, 0, 0,  1);

    // fpre_row / fpre_col: 32->64 with 1x3 / 3x1 slices of mw1_w (+bias)
    conv32_wmma<1, 3, 0><<<1024, 128, 0, stream>>>(feature, mw1w, mw1b, fprer,
                                                   4, 297, 9, 3, 3, 1);
    conv32_wmma<3, 1, 0><<<1024, 128, 0, stream>>>(feature, mw1w, mw1b, fprec,
                                                   4, 297, 9, 1, 3, 1);

    int nz = 4 * DHW;
    zero_kernel<<<(nz + 255) / 256, 256, 0, stream>>>(bufA, nz);

    for (int it = 0; it < 5; ++it) {  // ping-pong; 5 iters => final in bufB
        float* src = (it & 1) ? bufB : bufA;
        float* dst = (it & 1) ? bufA : bufB;
        float* s_mu = src;            float* s_md = src + DHW;
        float* s_ml = src + 2 * DHW;  float* s_mr = src + 3 * DHW;
        float* n_mu = dst;            float* n_md = dst + DHW;
        float* n_ml = dst + 2 * DHW;  float* n_mr = dst + 3 * DHW;
        msg_kernel<<<ND * H, 128, 0, stream>>>(cost, s_ml, s_mr, fprer, mw1w, mw2w,
                                               mw2b, sw_u, sw_d, n_mu, n_md, 1);
        msg_kernel<<<ND * W, 128, 0, stream>>>(cost, s_mu, s_md, fprec, mw1w, mw2w,
                                               mw2b, sw_l, sw_r, n_ml, n_mr, 0);
    }

    belief_kernel<<<ND * 1024, 128, 0, stream>>>(bufB, bw1w, bw1b, bw2w, bw2b,
                                                 (float*)d_out);
}
